// PSA_18940805775471
// MI455X (gfx1250) — matrix operand order repaired and checked by reading, not compiled, not run on hardware
//
#include <hip/hip_runtime.h>

// ---------------------------------------------------------------- types
typedef __bf16 bf16_t;
typedef __attribute__((ext_vector_type(16))) __bf16 v16bf;
typedef __attribute__((ext_vector_type(8)))  float  v8f;

union Frag {
  v16bf v;
  unsigned short u[16];
  uint4 q[2];
};

__device__ __forceinline__ unsigned short f2bf(float f) {
  unsigned int x = __float_as_uint(f);
  x += 0x7FFFu + ((x >> 16) & 1u);          // round-to-nearest-even
  return (unsigned short)(x >> 16);
}
__device__ __forceinline__ float bf2f(unsigned short u) {
  return __uint_as_float((unsigned int)u << 16);
}
__device__ __forceinline__ unsigned int cvt2(float a, float b) {
  return (unsigned)f2bf(a) | ((unsigned)f2bf(b) << 16);
}

// CDNA5 async copy: global -> LDS, 16B per lane, tracked by ASYNCcnt
__device__ __forceinline__ void async_copy16(unsigned lds_addr, const void* gaddr) {
  asm volatile("global_load_async_to_lds_b128 %0, %1, off"
               :: "v"(lds_addr), "v"(gaddr) : "memory");
}
#define WAIT_ASYNC(n) asm volatile("s_wait_asynccnt " #n ::: "memory")

__device__ __forceinline__ float hmax16(float v) {
  #pragma unroll
  for (int m = 1; m < 16; m <<= 1) v = fmaxf(v, __shfl_xor(v, m, 32));
  return v;
}
__device__ __forceinline__ float hsum16(float v) {
  #pragma unroll
  for (int m = 1; m < 16; m <<= 1) v += __shfl_xor(v, m, 32);
  return v;
}

// ------------------------------------------------ f32 -> bf16 (weights, x)
__global__ __launch_bounds__(256) void cvt_bf16(
    const float* __restrict__ src, unsigned short* __restrict__ dst, int n)
{
  int i = (blockIdx.x * 256 + threadIdx.x) * 2;
  if (i >= n) return;
  *(unsigned int*)&dst[i] = cvt2(src[i], src[i + 1]);
}

// ---------------------------------------------------------------- GEMM
// Y[b][m][n] = epilogue( sum_k W[m][k] * X[b][k][n] )
// W, X bf16; Y bf16 (f32out=0) or f32 (f32out=1). Double-buffered async LDS.
#define TB   256
#define BMT  128
#define BNT  128
#define BKT  32

__global__ __launch_bounds__(TB) void gemm_wmma(
    const unsigned short* __restrict__ W, const unsigned short* __restrict__ X,
    void* __restrict__ Y,
    const float* __restrict__ sc, const float* __restrict__ bi,
    int M, int N, int K, int act, int res, int f32out,
    size_t x_bstride, size_t y_bstride)
{
  __shared__ unsigned short As[2][BMT][BKT + 8];   // [m][k]
  __shared__ unsigned short Bs[2][BKT][BNT + 8];   // [k][n]

  const int b  = blockIdx.z;
  const int m0 = blockIdx.y * BMT;
  const int n0 = blockIdx.x * BNT;
  const unsigned short* Xb = X + (size_t)b * x_bstride;
  unsigned short* Yh = (unsigned short*)Y + (size_t)b * y_bstride;
  float*          Yf = (float*)Y          + (size_t)b * y_bstride;

  const int tid  = threadIdx.x;
  const int lane = tid & 31;
  const int wave = tid >> 5;
  const int wm0  = (wave & 3) * 32;   // 4 M-slots of 32
  const int wn0  = (wave >> 2) * 64;  // 2 N-slots of 64
  const int hid  = lane >> 4;
  const int mr   = lane & 15;
  const int c0   = hid * 8;

  v8f acc[2][4] = {};

  // async-stage one K-tile (4 instructions per wave)
  auto stageAB = [&](int buf, int k0) {
    #pragma unroll
    for (int j = 0; j < 2; ++j) {
      int idx = tid + j * TB;            // 0..511
      int mm  = idx >> 2;
      int kk  = (idx & 3) * 8;
      async_copy16((unsigned)(size_t)&As[buf][mm][kk],
                   &W[(size_t)(m0 + mm) * K + k0 + kk]);
    }
    #pragma unroll
    for (int j = 0; j < 2; ++j) {
      int idx = tid + j * TB;
      int kk  = idx >> 4;
      int nn  = (idx & 15) * 8;
      async_copy16((unsigned)(size_t)&Bs[buf][kk][nn],
                   &Xb[(size_t)(k0 + kk) * N + n0 + nn]);
    }
  };

  stageAB(0, 0);
  const int nk = K / BKT;
  for (int it = 0; it < nk; ++it) {
    const int cur = it & 1;
    if (it + 1 < nk) {
      stageAB(cur ^ 1, (it + 1) * BKT);  // overlap next tile with this one
      WAIT_ASYNC(0x4);                   // in-order: current tile resident
    } else {
      WAIT_ASYNC(0x0);
    }
    __syncthreads();

    Frag a[2];
    #pragma unroll
    for (int t = 0; t < 2; ++t) {
      const unsigned short* row = &As[cur][wm0 + t * 16 + mr][0];
      a[t].q[0] = *(const uint4*)(row + c0);
      a[t].q[1] = *(const uint4*)(row + 16 + c0);
    }
    #pragma unroll
    for (int nt = 0; nt < 4; ++nt) {
      Frag bf;
      const unsigned short* col = &Bs[cur][0][wn0 + nt * 16 + mr];
      #pragma unroll
      for (int s = 0; s < 16; ++s)
        bf.u[s] = col[(s < 8 ? c0 + s : 16 + c0 + s - 8) * (BNT + 8)];
      #pragma unroll
      for (int t = 0; t < 2; ++t)
        acc[t][nt] = __builtin_amdgcn_wmma_f32_16x16x32_bf16(
            false, a[t].v, false, bf.v, (short)0, acc[t][nt], false, false);
    }
    __syncthreads();
  }

  // fused epilogue (fp32 math, bf16 or f32 store)
  #pragma unroll
  for (int t = 0; t < 2; ++t) {
    #pragma unroll
    for (int r = 0; r < 8; ++r) {
      int m = m0 + wm0 + t * 16 + hid * 8 + r;
      float scl = sc[m], bia = bi[m];
      #pragma unroll
      for (int nt = 0; nt < 4; ++nt) {
        int n = n0 + wn0 + nt * 16 + mr;
        float v = acc[t][nt][r] * scl + bia;
        if (act) v = v * (1.0f / (1.0f + __expf(-v)));   // SiLU
        size_t o = (size_t)m * N + n;
        if (res) v += bf2f(Yh[o]);
        if (f32out) Yf[o] = v; else Yh[o] = f2bf(v);
      }
    }
  }
}

// ---------------------------------------------------------------- attention
// Flash-style: per (qblock=128, head, batch). qkv (bf16) per (b,h):
//   q at ch 0..31, k at 32..63, v at 64..127, each [ch][n=1024].
#define N_TOK 1024
#define KT    64

__global__ __launch_bounds__(256) void attn_wmma(
    const unsigned short* __restrict__ qkv, unsigned short* __restrict__ xa)
{
  __shared__ unsigned short Qs[128][32 + 8];   // [q][d]  (A layout source)
  __shared__ unsigned short Ks[32][KT + 8];    // [d][m]  (B for S)
  __shared__ unsigned short Vs[KT][64 + 8];    // [m][dv] (B for O)
  __shared__ unsigned short Ps[8][16][KT];     // per-wave P relayout

  const int qb0 = blockIdx.x * 128;
  const int h   = blockIdx.y;
  const int b   = blockIdx.z;
  const unsigned short* qp = qkv + (((size_t)b * 4 + h) * 128 + 0)  * N_TOK;
  const unsigned short* kp = qkv + (((size_t)b * 4 + h) * 128 + 32) * N_TOK;
  const unsigned short* vp = qkv + (((size_t)b * 4 + h) * 128 + 64) * N_TOK;

  const int tid  = threadIdx.x;
  const int lane = tid & 31;
  const int wave = tid >> 5;
  const int hid  = lane >> 4;
  const int mr   = lane & 15;
  const int c0   = hid * 8;
  const float scale = 0.17677669529663687f;   // 32^-0.5

  // stage Q transposed (pure bf16 scatter, no conversion)
  #pragma unroll
  for (int j = 0; j < 2; ++j) {
    int idx = tid + j * 256;                // 512 chunks of 8 queries
    int d   = idx >> 4;
    int qc  = (idx & 15) * 8;
    uint4 qv = *(const uint4*)&qp[(size_t)d * N_TOK + qb0 + qc];
    Qs[qc + 0][d] = (unsigned short)(qv.x & 0xffff);
    Qs[qc + 1][d] = (unsigned short)(qv.x >> 16);
    Qs[qc + 2][d] = (unsigned short)(qv.y & 0xffff);
    Qs[qc + 3][d] = (unsigned short)(qv.y >> 16);
    Qs[qc + 4][d] = (unsigned short)(qv.z & 0xffff);
    Qs[qc + 5][d] = (unsigned short)(qv.z >> 16);
    Qs[qc + 6][d] = (unsigned short)(qv.w & 0xffff);
    Qs[qc + 7][d] = (unsigned short)(qv.w >> 16);
  }
  __syncthreads();

  Frag qf;  // this wave's 16 query rows, K = full 32 dims
  {
    const unsigned short* row = &Qs[wave * 16 + mr][0];
    qf.q[0] = *(const uint4*)(row + c0);
    qf.q[1] = *(const uint4*)(row + 16 + c0);
  }

  v8f  o[4] = {};
  float rmax[8], rsum[8];
  #pragma unroll
  for (int r = 0; r < 8; ++r) { rmax[r] = -INFINITY; rsum[r] = 0.f; }

  for (int m0 = 0; m0 < N_TOK; m0 += KT) {
    __syncthreads();   // prior tile fully consumed
    // stage K tile [d][m] via async engine (1 instr/wave)
    {
      int d  = tid >> 3;
      int mc = (tid & 7) * 8;
      async_copy16((unsigned)(size_t)&Ks[d][mc], &kp[(size_t)d * N_TOK + m0 + mc]);
    }
    // stage V tile transposed [m][dv] (bf16 scatter)
    #pragma unroll
    for (int j = 0; j < 2; ++j) {
      int idx = tid + j * 256;
      int dv  = idx >> 3;
      int mc  = (idx & 7) * 8;
      uint4 vv = *(const uint4*)&vp[(size_t)dv * N_TOK + m0 + mc];
      Vs[mc + 0][dv] = (unsigned short)(vv.x & 0xffff);
      Vs[mc + 1][dv] = (unsigned short)(vv.x >> 16);
      Vs[mc + 2][dv] = (unsigned short)(vv.y & 0xffff);
      Vs[mc + 3][dv] = (unsigned short)(vv.y >> 16);
      Vs[mc + 4][dv] = (unsigned short)(vv.z & 0xffff);
      Vs[mc + 5][dv] = (unsigned short)(vv.z >> 16);
      Vs[mc + 6][dv] = (unsigned short)(vv.w & 0xffff);
      Vs[mc + 7][dv] = (unsigned short)(vv.w >> 16);
    }
    WAIT_ASYNC(0x0);
    __syncthreads();

    // S = Q^T K  (16 queries x 64 keys per wave), then scale
    v8f s[4];
    #pragma unroll
    for (int nt = 0; nt < 4; ++nt) {
      Frag kf;
      const unsigned short* col = &Ks[0][nt * 16 + mr];
      #pragma unroll
      for (int s = 0; s < 16; ++s)
        kf.u[s] = col[(s < 8 ? c0 + s : 16 + c0 + s - 8) * (KT + 8)];
      v8f z = {};
      s[nt] = __builtin_amdgcn_wmma_f32_16x16x32_bf16(
          false, qf.v, false, kf.v, (short)0, z, false, false);
      #pragma unroll
      for (int e = 0; e < 8; ++e) s[nt][e] *= scale;
    }

    // online softmax per row (half-wave reductions)
    #pragma unroll
    for (int r = 0; r < 8; ++r) {
      float tm = fmaxf(fmaxf(s[0][r], s[1][r]), fmaxf(s[2][r], s[3][r]));
      tm = hmax16(tm);
      float nm   = fmaxf(rmax[r], tm);
      float corr = __expf(rmax[r] - nm);
      rmax[r] = nm;
      float ps = 0.f;
      #pragma unroll
      for (int nt = 0; nt < 4; ++nt) {
        float p = __expf(s[nt][r] - nm);
        ps += p;
        Ps[wave][r + hid * 8][nt * 16 + mr] = f2bf(p);
      }
      ps = hsum16(ps);
      rsum[r] = rsum[r] * corr + ps;
      #pragma unroll
      for (int nt = 0; nt < 4; ++nt) o[nt][r] *= corr;
    }

    // same-wave LDS relayout: DS ops are in-order per wave; stop compiler reordering
    asm volatile("" ::: "memory");

    // O += P * V^T
    #pragma unroll
    for (int kc = 0; kc < 2; ++kc) {
      Frag pf;
      const unsigned short* prow = &Ps[wave][mr][kc * 32];
      pf.q[0] = *(const uint4*)(prow + c0);
      pf.q[1] = *(const uint4*)(prow + 16 + c0);
      #pragma unroll
      for (int nt = 0; nt < 4; ++nt) {
        Frag vf;
        const unsigned short* vcol = &Vs[kc * 32][nt * 16 + mr];
        #pragma unroll
        for (int s = 0; s < 16; ++s)
          vf.u[s] = vcol[(s < 8 ? c0 + s : 16 + c0 + s - 8) * (64 + 8)];
        o[nt] = __builtin_amdgcn_wmma_f32_16x16x32_bf16(
            false, pf.v, false, vf.v, (short)0, o[nt], false, false);
      }
    }
  }

  // normalize + store  xa[b][h*64+dv][q]  (bf16)
  #pragma unroll
  for (int nt = 0; nt < 4; ++nt) {
    int dv = nt * 16 + mr;
    unsigned short* op = &xa[(((size_t)b * 4 + h) * 64 + dv) * N_TOK];
    #pragma unroll
    for (int r = 0; r < 8; ++r) {
      int q = qb0 + wave * 16 + hid * 8 + r;
      op[q] = f2bf(o[nt][r] / rsum[r]);
    }
  }
}

// ------------------------------------------------ depthwise 3x3 PE + add
__global__ __launch_bounds__(256) void pe_add(
    const unsigned short* __restrict__ qkv, const float* __restrict__ Wd,
    const float* __restrict__ sc, const float* __restrict__ bi,
    unsigned short* __restrict__ xa)
{
  int idx = blockIdx.x * 256 + threadIdx.x;
  if (idx >= 16 * 256 * 1024) return;
  int p = idx & 1023;
  int c = (idx >> 10) & 255;
  int b = idx >> 18;
  int yy = p >> 5, xx = p & 31;
  int h = c >> 6, dv = c & 63;
  const unsigned short* vp = qkv + (((size_t)b * 4 + h) * 128 + 64 + dv) * 1024;
  float acc = 0.f;
  #pragma unroll
  for (int ky = -1; ky <= 1; ++ky) {
    int y2 = yy + ky;
    if (y2 < 0 || y2 > 31) continue;
    #pragma unroll
    for (int kx = -1; kx <= 1; ++kx) {
      int x2 = xx + kx;
      if (x2 < 0 || x2 > 31) continue;
      acc += Wd[c * 9 + (ky + 1) * 3 + (kx + 1)] * bf2f(vp[y2 * 32 + x2]);
    }
  }
  size_t o = ((size_t)b * 256 + c) * 1024 + p;
  xa[o] = f2bf(bf2f(xa[o]) + acc * sc[c] + bi[c]);
}

// ---------------------------------------------------------------- launch
extern "C" void kernel_launch(void* const* d_in, const int* in_sizes, int n_in,
                              void* d_out, int out_size, void* d_ws, size_t ws_size,
                              hipStream_t stream) {
  (void)in_sizes; (void)n_in; (void)out_size; (void)ws_size;
  const float* x      = (const float*)d_in[0];
  const float* W_cv1  = (const float*)d_in[1];
  const float* s_cv1  = (const float*)d_in[2];
  const float* b_cv1  = (const float*)d_in[3];
  const float* W_qkv  = (const float*)d_in[4];
  const float* s_qkv  = (const float*)d_in[5];
  const float* b_qkv  = (const float*)d_in[6];
  const float* W_proj = (const float*)d_in[7];
  const float* s_proj = (const float*)d_in[8];
  const float* b_proj = (const float*)d_in[9];
  const float* W_pe   = (const float*)d_in[10];
  const float* s_pe   = (const float*)d_in[11];
  const float* b_pe   = (const float*)d_in[12];
  const float* W_ffn1 = (const float*)d_in[13];
  const float* s_ffn1 = (const float*)d_in[14];
  const float* b_ffn1 = (const float*)d_in[15];
  const float* W_ffn2 = (const float*)d_in[16];
  const float* s_ffn2 = (const float*)d_in[17];
  const float* b_ffn2 = (const float*)d_in[18];
  const float* W_cv2  = (const float*)d_in[19];
  const float* s_cv2  = (const float*)d_in[20];
  const float* b_cv2  = (const float*)d_in[21];
  float* out = (float*)d_out;

  // all activations bf16 in workspace (fits easily in 192MB L2)
  unsigned short* wsh  = (unsigned short*)d_ws;
  unsigned short* ycat = wsh;                  // [16][512][1024]  a||bb
  unsigned short* qkvb = wsh + 8388608;        // [16][512][1024]  qkv / ffn hidden
  unsigned short* xab  = wsh + 16777216;       // [16][256][1024]
  unsigned short* xbf  = wsh + 20971520;       // [16][512][1024]  x in bf16
  unsigned short* wbf  = wsh + 29360128;       // weights
  unsigned short* bb   = ycat + 256 * 1024;    // bb view (batch stride 512*1024)

  unsigned short* wb_cv1  = wbf;               // 512*512
  unsigned short* wb_qkv  = wbf + 262144;      // 512*256
  unsigned short* wb_proj = wbf + 393216;      // 256*256
  unsigned short* wb_ffn1 = wbf + 458752;      // 512*256
  unsigned short* wb_ffn2 = wbf + 589824;      // 256*512
  unsigned short* wb_cv2  = wbf + 720896;      // 512*512

  const size_t S512 = 512 * 1024, S256 = 256 * 1024;
  dim3 blk(TB);

  // 0. one-time conversions (weights + input x)
  cvt_bf16<<<dim3(512),   dim3(256), 0, stream>>>(W_cv1,  wb_cv1,  262144);
  cvt_bf16<<<dim3(256),   dim3(256), 0, stream>>>(W_qkv,  wb_qkv,  131072);
  cvt_bf16<<<dim3(128),   dim3(256), 0, stream>>>(W_proj, wb_proj,  65536);
  cvt_bf16<<<dim3(256),   dim3(256), 0, stream>>>(W_ffn1, wb_ffn1, 131072);
  cvt_bf16<<<dim3(256),   dim3(256), 0, stream>>>(W_ffn2, wb_ffn2, 131072);
  cvt_bf16<<<dim3(512),   dim3(256), 0, stream>>>(W_cv2,  wb_cv2,  262144);
  cvt_bf16<<<dim3(16384), dim3(256), 0, stream>>>(x,      xbf,   8388608);

  // 1. y = SiLU(cv1(x)) -> ycat (a||bb)
  gemm_wmma<<<dim3(8, 4, 16), blk, 0, stream>>>(wb_cv1, xbf, ycat, s_cv1, b_cv1,
      512, 1024, 512, 1, 0, 0, S512, S512);
  // 2. qkv = cv(bb)
  gemm_wmma<<<dim3(8, 4, 16), blk, 0, stream>>>(wb_qkv, bb, qkvb, s_qkv, b_qkv,
      512, 1024, 256, 0, 0, 0, S512, S512);
  // 3. flash attention -> xab
  attn_wmma<<<dim3(8, 4, 16), dim3(256), 0, stream>>>(qkvb, xab);
  // 4. xab += dwconv3x3(v)*s+b
  pe_add<<<dim3(16384), dim3(256), 0, stream>>>(qkvb, W_pe, s_pe, b_pe, xab);
  // 5. bb += proj(xab)
  gemm_wmma<<<dim3(8, 2, 16), blk, 0, stream>>>(wb_proj, xab, bb, s_proj, b_proj,
      256, 1024, 256, 0, 1, 0, S256, S512);
  // 6. ff = SiLU(ffn1(bb)) -> qkvb (reused)
  gemm_wmma<<<dim3(8, 4, 16), blk, 0, stream>>>(wb_ffn1, bb, qkvb, s_ffn1, b_ffn1,
      512, 1024, 256, 1, 0, 0, S512, S512);
  // 7. bb += ffn2(ff)
  gemm_wmma<<<dim3(8, 2, 16), blk, 0, stream>>>(wb_ffn2, qkvb, bb, s_ffn2, b_ffn2,
      256, 1024, 512, 0, 1, 0, S512, S512);
  // 8. out = SiLU(cv2(a||bb))  (f32 to d_out)
  gemm_wmma<<<dim3(8, 4, 16), blk, 0, stream>>>(wb_cv2, ycat, out, s_cv2, b_cv2,
      512, 1024, 512, 1, 0, 1, S512, S512);
}
